// Model_20547123544422
// MI455X (gfx1250) — matrix operand order repaired and checked by reading, not compiled, not run on hardware
//
#include <hip/hip_runtime.h>
#include <hip/hip_bf16.h>

// ---------------- problem constants ----------------
#define H_   256
#define L_   6
#define IN_  128
#define B_   128
#define T_   512
#define NWG  16       // persistent workgroups for the recurrent scan
#define KMAX 512      // max GEMM K (= H + Din)

typedef __attribute__((ext_vector_type(16))) __bf16 v16bf;
typedef __attribute__((ext_vector_type(8)))  float  v8f;
typedef __attribute__((ext_vector_type(4)))  int    v4i;

#define AS1 __attribute__((address_space(1)))
#define AS3 __attribute__((address_space(3)))

#if __has_builtin(__builtin_amdgcn_global_load_async_to_lds_b128) && \
    __has_builtin(__builtin_amdgcn_s_wait_asynccnt)
#define USE_ASYNC_LDS 1
#else
#define USE_ASYNC_LDS 0
#endif

// ---------------- helpers ----------------
__device__ __forceinline__ unsigned short f2bf(float f) {
  union { float f; unsigned u; } v; v.f = f;
  unsigned u = v.u;
  u += 0x7fffu + ((u >> 16) & 1u);   // round-to-nearest-even
  return (unsigned short)(u >> 16);
}
__device__ __forceinline__ float fast_sig(float x) {
  float e = __builtin_amdgcn_exp2f(-1.4426950408889634f * x);
  return __builtin_amdgcn_rcpf(1.0f + e);
}
__device__ __forceinline__ float fast_tanh(float x) {
  float e = __builtin_amdgcn_exp2f(-2.8853900817779268f * x);  // exp2(-2x*log2e)
  return 2.0f * __builtin_amdgcn_rcpf(1.0f + e) - 1.0f;
}

// ---------------- prep kernels ----------------
__global__ void zero_bar(unsigned* bar) { if (threadIdx.x == 0) *bar = 0u; }

__global__ void prep_x(const float* __restrict__ x, unsigned short* __restrict__ xbf, int n) {
  int i = blockIdx.x * blockDim.x + threadIdx.x;
  if (i < n) xbf[i] = f2bf(x[i]);
}

__global__ void prep_bias(const float* __restrict__ b_ih, const float* __restrict__ b_hh,
                          float* __restrict__ bias) {
  int i = blockIdx.x * blockDim.x + threadIdx.x;   // L*4H
  bias[i] = b_ih[i] + b_hh[i];
}

// Pack per-layer weights bf16 row-major: row r in [0,4H), cols [0,H)=W_hh, cols [H,K)=W_ih
__global__ void prep_w(const float* __restrict__ w_ih0, const float* __restrict__ w_ih,
                       const float* __restrict__ w_hh, unsigned short* __restrict__ wc) {
  int l   = blockIdx.y;
  int idx = blockIdx.x * blockDim.x + threadIdx.x;   // 0 .. 1024*512
  int row = idx >> 9;
  int col = idx & 511;
  int K = (l == 0) ? (H_ + IN_) : (2 * H_);
  if (col >= K) return;
  float v;
  if (col < H_) {
    v = w_hh[(size_t)(l * 4 * H_ + row) * H_ + col];
  } else {
    int d = col - H_;
    v = (l == 0) ? w_ih0[(size_t)row * IN_ + d]
                 : w_ih[(size_t)((l - 1) * 4 * H_ + row) * H_ + d];
  }
  size_t off = (l == 0) ? 0 : (size_t)1024 * 384 + (size_t)(l - 1) * 1024 * 512;
  wc[off + (size_t)row * K + col] = f2bf(v);
}

// ---------------- persistent recurrent-scan kernel (one layer) ----------------
__global__ __launch_bounds__(256) void lstm_layer(
    const unsigned short* __restrict__ xin,   // [B,T,Din] bf16
    unsigned short* __restrict__ yout,        // [B,T,H] bf16
    const unsigned short* __restrict__ wc,    // [4H,K] bf16 packed [Whh|Wih]
    const float* __restrict__ bias,           // [4H] (b_ih + b_hh)
    const float* __restrict__ h0,             // [B,H] layer slice, f32
    const float* __restrict__ c0,             // [B,H]
    float* __restrict__ hout,                 // d_out + l*B*H
    unsigned short* __restrict__ hbuf,        // 2 * B*H bf16 ping-pong
    unsigned* __restrict__ bar,
    int Din, unsigned barBase)
{
  const int K    = H_ + Din;
  const int tid  = threadIdx.x;
  const int wave = tid >> 5;
  const int lane = tid & 31;
  const int ln   = lane & 15;
  const int sel  = lane >> 4;
  const int j0   = blockIdx.x * 16;   // hidden-column strip
  const int m0   = wave * 16;         // batch-row tile

  // 64 rows (4 gates x 16 hidden cols) x K bf16, XOR-swizzled 16B chunks
  __shared__ unsigned short lds[64 * KMAX];   // 64KB

  // ---- stage weight slice into LDS in 16B chunks (async-to-LDS when available) ----
  {
    const int cpr = K >> 3;            // 16B chunks per row
    const int nch = 64 * cpr;
    for (int idx = tid; idx < nch; idx += 256) {
      int r  = idx / cpr;
      int c  = idx - r * cpr;
      int g = r >> 4, n = r & 15;
      int cs = c ^ (r & 15);
      const unsigned short* src = wc + (size_t)(g * H_ + j0 + n) * K + (c << 3);
      unsigned short*       dst = &lds[r * K + (cs << 3)];
#if USE_ASYNC_LDS
      __builtin_amdgcn_global_load_async_to_lds_b128(
          (AS1 v4i*)(size_t)src, (AS3 v4i*)(size_t)dst, 0, 0);
#else
      *(uint4*)dst = *(const uint4*)src;
#endif
    }
#if USE_ASYNC_LDS
    __builtin_amdgcn_s_wait_asynccnt(0);
#endif
  }

  // per-lane biases (column = j0+ln fixed for this lane)
  float bia[4];
#pragma unroll
  for (int g = 0; g < 4; ++g) bia[g] = bias[g * H_ + j0 + ln];

  // cell state lives in registers; seed h_prev (buffer 0) from h0
  float c_r[8];
  const int rbase = m0 + sel * 8;     // C/D layout: VGPR r -> M = rbase + r
#pragma unroll
  for (int r = 0; r < 8; ++r) {
    int brow = rbase + r, j = j0 + ln;
    c_r[r] = c0[brow * H_ + j];
    hbuf[brow * H_ + j] = f2bf(h0[brow * H_ + j]);
  }

  unsigned barIdx = barBase;
  auto gbar = [&]() {
    __syncthreads();
    __threadfence();
    ++barIdx;
    if (tid == 0) {
      atomicAdd(bar, 1u);
      while (__hip_atomic_load(bar, __ATOMIC_ACQUIRE, __HIP_MEMORY_SCOPE_AGENT) <
             barIdx * (unsigned)NWG)
        __builtin_amdgcn_s_sleep(1);
    }
    __syncthreads();
  };

  gbar();   // h_prev + LDS weights visible everywhere

  const unsigned short* hprev = hbuf;
  unsigned short*       hcur  = hbuf + B_ * H_;

  const int arow = m0 + ln;           // A layout: lane -> M = m0 + (lane&15)
  const unsigned short* xrow = xin + (size_t)arow * T_ * Din + sel * 8;

  for (int t = 0; t < T_; ++t) {
    v8f acc[4];
#pragma unroll
    for (int g = 0; g < 4; ++g) {
      v8f a;
#pragma unroll
      for (int r = 0; r < 8; ++r) a[r] = bia[g];
      acc[g] = a;
    }

    // prefetch next timestep's input rows while we compute this step
    if (t + 1 < T_) {
      const unsigned short* pf = xin + ((size_t)arow * T_ + (t + 1)) * Din;
      __builtin_prefetch(pf, 0, 1);
      __builtin_prefetch(pf + 64, 0, 1);
    }

    // D = [h_{t-1} | x_t] x [Whh | Wih]^T  accumulated per gate
    auto mm4 = [&](const v16bf& Av, int kb) {
      int cbase = (kb >> 3) + sel;
#pragma unroll
      for (int g = 0; g < 4; ++g) {
        int r = g * 16 + ln;
        const unsigned short* lp = &lds[r * K];
        union { v16bf v; uint4 q[2]; } Bm;
        Bm.q[0] = *(const uint4*)(lp + (((cbase    ) ^ (r & 15)) << 3));
        Bm.q[1] = *(const uint4*)(lp + (((cbase + 2) ^ (r & 15)) << 3));
        acc[g] = __builtin_amdgcn_wmma_f32_16x16x32_bf16(
            false, Av, false, Bm.v, (short)0, acc[g], false, false);
      }
    };

    // recurrent part: K = 0..255 from h_{t-1}
#pragma unroll 2
    for (int kk = 0; kk < (H_ >> 5); ++kk) {
      int kb = kk << 5;
      union { v16bf v; uint4 q[2]; } A;
      const unsigned short* p = hprev + arow * H_ + kb + sel * 8;
      A.q[0] = *(const uint4*)p;
      A.q[1] = *(const uint4*)(p + 16);
      mm4(A.v, kb);
    }
    // input part: K = 256.. from x_t
    const unsigned short* xp = xrow + (size_t)t * Din;
    const int nkx = Din >> 5;
#pragma unroll 2
    for (int kx = 0; kx < nkx; ++kx) {
      int kb = H_ + (kx << 5);
      union { v16bf v; uint4 q[2]; } A;
      const unsigned short* p = xp + (kx << 5);
      A.q[0] = *(const uint4*)p;
      A.q[1] = *(const uint4*)(p + 16);
      mm4(A.v, kb);
    }

    // gate math (all four gates for this (b,j) block are lane-local)
    const int j = j0 + ln;
    const bool last = (t == T_ - 1);
#pragma unroll
    for (int r = 0; r < 8; ++r) {
      int brow = rbase + r;
      float ig = fast_sig(acc[0][r]);
      float fg = fast_sig(acc[1][r]);
      float gg = fast_tanh(acc[2][r]);
      float og = fast_sig(acc[3][r]);
      float cc = fg * c_r[r] + ig * gg;
      c_r[r] = cc;
      float hh = og * fast_tanh(cc);
      unsigned short hb = f2bf(hh);
      hcur[brow * H_ + j] = hb;
      yout[((size_t)brow * T_ + t) * H_ + j] = hb;
      if (last) hout[brow * H_ + j] = hh;
    }
    gbar();
    unsigned short* tmp = hcur;
    hcur  = (unsigned short*)hprev;
    hprev = tmp;
  }
}

// ---------------- host launcher ----------------
extern "C" void kernel_launch(void* const* d_in, const int* in_sizes, int n_in,
                              void* d_out, int out_size, void* d_ws, size_t ws_size,
                              hipStream_t stream) {
  const float* x     = (const float*)d_in[0];
  const float* h0    = (const float*)d_in[1];
  const float* c0    = (const float*)d_in[2];
  const float* w_ih0 = (const float*)d_in[3];
  const float* w_ih  = (const float*)d_in[4];
  const float* w_hh  = (const float*)d_in[5];
  const float* b_ih  = (const float*)d_in[6];
  const float* b_hh  = (const float*)d_in[7];
  float* out = (float*)d_out;

  char* ws = (char*)d_ws;
  unsigned*       bar  = (unsigned*)ws;                              // 256 B
  float*          bias = (float*)(ws + 256);                         // 24 KB
  unsigned short* wc   = (unsigned short*)(ws + 256 + 24576);
  size_t wcBytes = 2ull * 1024 * (384 + 5 * 512);                    // ~5.9 MB
  unsigned short* hbuf = (unsigned short*)((char*)wc + wcBytes);     // 128 KB
  unsigned short* xbf  = (unsigned short*)((char*)hbuf + (size_t)2 * B_ * H_ * 2);
  unsigned short* y0   = (unsigned short*)((char*)xbf + (size_t)B_ * T_ * IN_ * 2);
  unsigned short* y1   = (unsigned short*)((char*)y0  + (size_t)B_ * T_ * H_ * 2);

  zero_bar<<<1, 32, 0, stream>>>(bar);
  prep_x<<<(B_ * T_ * IN_) / 256, 256, 0, stream>>>(x, xbf, B_ * T_ * IN_);
  prep_bias<<<(L_ * 4 * H_) / 256, 256, 0, stream>>>(b_ih, b_hh, bias);
  prep_w<<<dim3((1024 * 512) / 256, L_), 256, 0, stream>>>(w_ih0, w_ih, w_hh, wc);

  for (int l = 0; l < L_; ++l) {
    int Din = (l == 0) ? IN_ : H_;
    const unsigned short* xin = (l == 0) ? xbf : ((l & 1) ? y0 : y1);
    unsigned short*       yo  = (l & 1) ? y1 : y0;
    size_t woff = (l == 0) ? 0 : (size_t)1024 * 384 + (size_t)(l - 1) * 1024 * 512;
    lstm_layer<<<NWG, 256, 0, stream>>>(
        xin, yo, wc + woff, bias + l * 4 * H_,
        h0 + (size_t)l * B_ * H_, c0 + (size_t)l * B_ * H_,
        out + (size_t)l * B_ * H_, hbuf, bar, Din, (unsigned)(l * (T_ + 1)));
  }
}
